// AttentionHead_12661563588898
// MI455X (gfx1250) — matrix operand order repaired and checked by reading, not compiled, not run on hardware
//
#include <hip/hip_runtime.h>

typedef __attribute__((ext_vector_type(16))) _Float16 v16h;
typedef __attribute__((ext_vector_type(8)))  _Float16 v8h;
typedef __attribute__((ext_vector_type(8)))  float    v8f;

#define B_   8
#define S_   2048
#define DIN_ 1024
#define D_   64

// workspace layout, in _Float16 units
#define WSWZ_ELEMS (32*3*4*32*16)     /* 196608: [kchunk][mat][ntile][lane][e] */
#define CHUNK_H    (3*4*32*16)        /* 6144 halves = 12KB per k-chunk */
#define QKV_ELEMS  (B_*S_*D_)         /* 1048576 */
#define Q_OFF   (WSWZ_ELEMS)
#define K_OFF   (Q_OFF + QKV_ELEMS)
#define VT_OFF  (K_OFF + QKV_ELEMS)

// ---------------------------------------------------------------------------
// CDNA5 async global->LDS staging (ASYNCcnt path), with full fallback to the
// proven register-staged path when the builtin is unavailable.
// Builtin prototype (from compiler diagnostic): first arg is a pointer to
// int __vector(4) in the global address space; LDS side is the AS(3) twin.
// ---------------------------------------------------------------------------
#if defined(__AMDGCN__) && __has_builtin(__builtin_amdgcn_global_load_async_to_lds_b128)
#define ASYNC_LDS 1
typedef int v4i __attribute__((vector_size(4 * sizeof(int))));
typedef __attribute__((address_space(1))) v4i as1_v4i;
typedef __attribute__((address_space(3))) v4i as3_v4i;
__device__ __forceinline__ void async_copy16(const _Float16* g, _Float16* l) {
    __builtin_amdgcn_global_load_async_to_lds_b128(
        (as1_v4i*)(void*)const_cast<_Float16*>(g),
        (as3_v4i*)(void*)l, 0, 0);
}
__device__ __forceinline__ void wait_async0() {
#if __has_builtin(__builtin_amdgcn_s_wait_asynccnt)
    __builtin_amdgcn_s_wait_asynccnt(0);
#else
    asm volatile("s_wait_asynccnt 0x0" ::: "memory");
#endif
}
#else
#define ASYNC_LDS 0
#endif

// ---------------------------------------------------------------------------
// Kernel 0: pre-swizzle Wq/Wk/Wv (f32, [1024,64]) into f16 B-fragment order,
// chunk-contiguous: [c][m][nt][lane][e].  B-frag mapping (16x16x32 f16 WMMA):
// lane L holds column n = L%16; element e covers K = 8*(L/16) + e (e<8), 16 + 8*(L/16) + e-8 (e>=8).
// ---------------------------------------------------------------------------
__global__ void wswz_kernel(const float* __restrict__ Wq,
                            const float* __restrict__ Wk,
                            const float* __restrict__ Wv,
                            _Float16* __restrict__ out)
{
    int idx = blockIdx.x * blockDim.x + threadIdx.x;
    if (idx >= WSWZ_ELEMS) return;
    int e    = idx & 15;
    int l    = (idx >> 4) & 31;
    int nt   = (idx >> 9) & 3;
    int mc   = idx >> 11;        // 0..95 = c*3 + m
    int m    = mc % 3;
    int c    = mc / 3;
    const float* W = (m == 0) ? Wq : ((m == 1) ? Wk : Wv);
    int half = l >> 4;
    int n16  = l & 15;
    int row  = c * 32 + ((e < 8) ? (half * 8 + e) : (16 + half * 8 + e - 8));    // K index within DIN
    int col  = nt * 16 + n16;             // N index within D
    out[idx] = (_Float16)W[row * D_ + col];
}

// ---------------------------------------------------------------------------
// Kernel 1: QKV projection. One wave per 16-row tile of flattened [B*S, DIN].
// Block of 8 waves shares each 12KB W chunk via double-buffered LDS; chunk
// c+1 streams in (async-to-LDS if available) while chunk c feeds 12 WMMAs.
// Writes Q,K row-major f16; V transposed ([b][d][s]) f16.
// ---------------------------------------------------------------------------
__global__ __launch_bounds__(256)
void qkv_kernel(const float* __restrict__ X,
                const _Float16* __restrict__ Wswz,
                const float* __restrict__ bq,
                const float* __restrict__ bk,
                const float* __restrict__ bv,
                _Float16* __restrict__ Qh,
                _Float16* __restrict__ Kh,
                _Float16* __restrict__ Vt)
{
    __shared__ _Float16 wlds[2][CHUNK_H];   // 2 x 12KB: [m][nt][lane][e]

    const int tid   = threadIdx.x;
    const int lane  = tid & 31;
    const int wline = tid >> 5;
    const int half  = lane >> 4;
    const int n16   = lane & 15;
    const long g0   = ((long)blockIdx.x * 8 + wline) * 16;

    const float* xrow = X + (g0 + n16) * DIN_;

    // staging: thread covers halves [tid*24, tid*24+24) of the 6144-half chunk
    const _Float16* wg = Wswz + tid * 24;

    v8f accq[4], acck[4], accv[4];
#pragma unroll
    for (int nt = 0; nt < 4; ++nt) { accq[nt] = {}; acck[nt] = {}; accv[nt] = {}; }

    // prologue prefetch (chunk 0)
#if ASYNC_LDS
    {
        _Float16* wd = &wlds[0][tid * 24];
        async_copy16(wg + 0,  wd + 0);
        async_copy16(wg + 8,  wd + 8);
        async_copy16(wg + 16, wd + 16);
    }
#else
    v8h w0 = *(const v8h*)(wg + 0);
    v8h w1 = *(const v8h*)(wg + 8);
    v8h w2 = *(const v8h*)(wg + 16);
#endif
    float4 xa0 = *(const float4*)(xrow + 8*half);
    float4 xa1 = *(const float4*)(xrow + 8*half + 4);
    float4 xa2 = *(const float4*)(xrow + 16 + 8*half);
    float4 xa3 = *(const float4*)(xrow + 16 + 8*half + 4);

    for (int c = 0; c < 32; ++c) {
        const int cur = c & 1;
#if ASYNC_LDS
        wait_async0();                 // my chunk-c transfers have landed
        __syncthreads();               // everyone's landed; buf[1-cur] readers done
        if (c + 1 < 32) {              // stream chunk c+1 while computing chunk c
            const _Float16* wn = wg + (c + 1) * CHUNK_H;
            _Float16* wd = &wlds[1 - cur][tid * 24];
            async_copy16(wn + 0,  wd + 0);
            async_copy16(wn + 8,  wd + 8);
            async_copy16(wn + 16, wd + 16);
        }
#else
        __syncthreads();               // previous chunk's compute done
        {
            _Float16* wd = &wlds[cur][tid * 24];
            *(v8h*)(wd + 0)  = w0;
            *(v8h*)(wd + 8)  = w1;
            *(v8h*)(wd + 16) = w2;
        }
        if (c + 1 < 32) {
            const _Float16* wn = wg + (c + 1) * CHUNK_H;
            w0 = *(const v8h*)(wn + 0);
            w1 = *(const v8h*)(wn + 8);
            w2 = *(const v8h*)(wn + 16);
        }
#endif
        // build A fragment from prefetched X
        v16h a;
        a[0]=(_Float16)xa0.x; a[1]=(_Float16)xa0.y; a[2]=(_Float16)xa0.z; a[3]=(_Float16)xa0.w;
        a[4]=(_Float16)xa1.x; a[5]=(_Float16)xa1.y; a[6]=(_Float16)xa1.z; a[7]=(_Float16)xa1.w;
        a[8]=(_Float16)xa2.x; a[9]=(_Float16)xa2.y; a[10]=(_Float16)xa2.z; a[11]=(_Float16)xa2.w;
        a[12]=(_Float16)xa3.x; a[13]=(_Float16)xa3.y; a[14]=(_Float16)xa3.z; a[15]=(_Float16)xa3.w;

        if (c + 1 < 32) {              // prefetch next X A-fragment
            const float* xn = xrow + (c + 1) * 32;
            xa0 = *(const float4*)(xn + 8*half);
            xa1 = *(const float4*)(xn + 8*half + 4);
            xa2 = *(const float4*)(xn + 16 + 8*half);
            xa3 = *(const float4*)(xn + 16 + 8*half + 4);
        }
#if !ASYNC_LDS
        __syncthreads();               // chunk visible in LDS
#endif
        const _Float16* wl = wlds[cur];
#pragma unroll
        for (int nt = 0; nt < 4; ++nt) {
            v16h fq = *(const v16h*)&wl[((0*4 + nt)*32 + lane)*16];
            v16h fk = *(const v16h*)&wl[((1*4 + nt)*32 + lane)*16];
            v16h fv = *(const v16h*)&wl[((2*4 + nt)*32 + lane)*16];
            accq[nt] = __builtin_amdgcn_wmma_f32_16x16x32_f16(false, a, false, fq, (short)0, accq[nt], false, false);
            acck[nt] = __builtin_amdgcn_wmma_f32_16x16x32_f16(false, a, false, fk, (short)0, acck[nt], false, false);
            accv[nt] = __builtin_amdgcn_wmma_f32_16x16x32_f16(false, a, false, fv, (short)0, accv[nt], false, false);
        }
    }

    // Epilogue: bias add + stores. C layout: reg r, lane -> (M = r + 8*half, N = n16)
    const int b  = (int)(g0 >> 11);
    const int sl = (int)(g0 & (S_ - 1));
#pragma unroll
    for (int nt = 0; nt < 4; ++nt) {
        const int col = nt * 16 + n16;
        const float bqv = bq[col], bkv = bk[col], bvv = bv[col];
#pragma unroll
        for (int r = 0; r < 8; ++r) {
            const long row = g0 + r + 8*half;
            Qh[row * D_ + col] = (_Float16)(accq[nt][r] + bqv);
            Kh[row * D_ + col] = (_Float16)(acck[nt][r] + bkv);
        }
        v8h hv;
#pragma unroll
        for (int r = 0; r < 8; ++r) hv[r] = (_Float16)(accv[nt][r] + bvv);
        // Vt[b][col][sl + 8*half + r]; 8 consecutive s -> one 16B store
        _Float16* vp = Vt + ((long)(b * D_ + col) * S_ + sl + 8*half);
        *(v8h*)vp = hv;
    }
}

// ---------------------------------------------------------------------------
// Kernel 2: streaming (flash) attention. 8 waves/block = 8 q-tiles of ONE
// batch; the block shares each 32-key K/V slab via double-buffered LDS with
// the next slab streaming in (async-to-LDS if available) during compute.
// ---------------------------------------------------------------------------
__global__ __launch_bounds__(256)
void attn_kernel(const _Float16* __restrict__ Qh,
                 const _Float16* __restrict__ Kh,
                 const _Float16* __restrict__ Vt,
                 const unsigned char* __restrict__ mask,
                 float* __restrict__ out)
{
    __shared__ _Float16 kslab[2][32 * 64];   // [key][d]   2 x 4KB
    __shared__ _Float16 vslab[2][64 * 32];   // [d][key]   2 x 4KB
    __shared__ _Float16 plds[8][16 * 32];    // per-wave P transpose, 8KB

    const int tid   = threadIdx.x;
    const int lane  = tid & 31;
    const int wline = tid >> 5;
    const int half  = lane >> 4;
    const int n16   = lane & 15;
    const int tile  = blockIdx.x * 8 + wline;   // block stays within one batch
    const int b     = tile >> 7;
    const int q0    = (tile & 127) * 16;

    const _Float16* Kb  = Kh + (long)b * S_ * D_;
    const _Float16* Vtb = Vt + (long)b * D_ * S_;
    const unsigned char* mb = mask + b * S_;

    // slab staging assignments (16B per thread per slab)
    const int krow = tid >> 3, kseg = tid & 7;   // K: 32 rows x 8 segs
    const int vd   = tid >> 2, vseg = tid & 3;   // V: 64 rows x 4 segs
    const _Float16* kgp = Kb  + krow * D_ + kseg * 8;      // + key0*D_
    const _Float16* vgp = Vtb + (long)vd * S_ + vseg * 8;  // + key0
    const int koff = krow * 64 + kseg * 8;
    const int voff = vd * 32 + vseg * 8;

    // Q tile A-fragments, loaded once (16 rows x 64 d = 2 chunks of 32)
    const _Float16* Qrow = Qh + ((long)b * S_ + q0 + n16) * D_;
    v16h aq[2];
#pragma unroll
    for (int c = 0; c < 2; ++c) {
        v8h lo = *(const v8h*)(Qrow + 32*c + 16*half);
        v8h hi = *(const v8h*)(Qrow + 32*c + 16*half + 8);
        aq[c] = __builtin_shufflevector(lo, hi, 0,1,2,3,4,5,6,7,8,9,10,11,12,13,14,15);
    }

    float mrun[8], lrun[8];
#pragma unroll
    for (int r = 0; r < 8; ++r) { mrun[r] = -3.0e38f; lrun[r] = 0.0f; }
    v8f ctx[4];
#pragma unroll
    for (int nt = 0; nt < 4; ++nt) ctx[nt] = {};

    const float inv_scale = 0.022097086912079608f;  // 1/sqrt(2048)

    // prologue prefetch (slab 0)
#if ASYNC_LDS
    async_copy16(kgp, &kslab[0][koff]);
    async_copy16(vgp, &vslab[0][voff]);
#else
    v8h kreg = *(const v8h*)(kgp);
    v8h vreg = *(const v8h*)(vgp);
#endif

    for (int key0 = 0; key0 < S_; key0 += 32) {
        const int cur = (key0 >> 5) & 1;
#if ASYNC_LDS
        wait_async0();                 // my slab transfers have landed
        __syncthreads();               // everyone's landed; buf[1-cur] readers done
        if (key0 + 32 < S_) {          // stream next slab during compute
            async_copy16(kgp + (key0 + 32) * D_, &kslab[1 - cur][koff]);
            async_copy16(vgp + (key0 + 32),      &vslab[1 - cur][voff]);
        }
#else
        __syncthreads();
        *(v8h*)&kslab[cur][koff] = kreg;
        *(v8h*)&vslab[cur][voff] = vreg;
        if (key0 + 32 < S_) {
            kreg = *(const v8h*)(kgp + (key0 + 32) * D_);
            vreg = *(const v8h*)(vgp + (key0 + 32));
        }
        __syncthreads();               // slab visible
#endif
        const _Float16* ksl = kslab[cur];
        const _Float16* vsl = vslab[cur];

        // ---- scores for two 16-key tiles: S = Q(16x64) @ K^T, from LDS ----
        v8f s0 = {}, s1 = {};
        {
            const _Float16* kp = &ksl[n16 * 64 + 16*half];
            v16h b0 = *(const v16h*)(kp);
            v16h b1 = *(const v16h*)(kp + 32);
            s0 = __builtin_amdgcn_wmma_f32_16x16x32_f16(false, aq[0], false, b0, (short)0, s0, false, false);
            s0 = __builtin_amdgcn_wmma_f32_16x16x32_f16(false, aq[1], false, b1, (short)0, s0, false, false);
        }
        {
            const _Float16* kp = &ksl[(16 + n16) * 64 + 16*half];
            v16h b0 = *(const v16h*)(kp);
            v16h b1 = *(const v16h*)(kp + 32);
            s1 = __builtin_amdgcn_wmma_f32_16x16x32_f16(false, aq[0], false, b0, (short)0, s1, false, false);
            s1 = __builtin_amdgcn_wmma_f32_16x16x32_f16(false, aq[1], false, b1, (short)0, s1, false, false);
        }

        // mask column (broadcast over query rows): 1 byte per key
        const bool mk0 = mb[key0 + n16] != 0;
        const bool mk1 = mb[key0 + 16 + n16] != 0;

        // ---- online softmax, row reductions across the 16-lane C groups ----
#pragma unroll
        for (int r = 0; r < 8; ++r) {
            float v0 = mk0 ? -1.0e9f : s0[r] * inv_scale;
            float v1 = mk1 ? -1.0e9f : s1[r] * inv_scale;
            float tmax = fmaxf(v0, v1);
            tmax = fmaxf(tmax, __shfl_xor(tmax, 1, 32));
            tmax = fmaxf(tmax, __shfl_xor(tmax, 2, 32));
            tmax = fmaxf(tmax, __shfl_xor(tmax, 4, 32));
            tmax = fmaxf(tmax, __shfl_xor(tmax, 8, 32));
            const float mnew = fmaxf(mrun[r], tmax);
            const float corr = __expf(mrun[r] - mnew);
            mrun[r] = mnew;
            const float p0 = __expf(v0 - mnew);
            const float p1 = __expf(v1 - mnew);
            float rs = p0 + p1;
            rs += __shfl_xor(rs, 1, 32);
            rs += __shfl_xor(rs, 2, 32);
            rs += __shfl_xor(rs, 4, 32);
            rs += __shfl_xor(rs, 8, 32);
            lrun[r] = lrun[r] * corr + rs;
#pragma unroll
            for (int nt = 0; nt < 4; ++nt) ctx[nt][r] *= corr;
            const int M = r + 8*half;
            plds[wline][M * 32 + n16]      = (_Float16)p0;
            plds[wline][M * 32 + 16 + n16] = (_Float16)p1;
        }
        __syncthreads();

        // ---- P (16x32) as A-fragment from LDS ----
        v16h ap;
        {
            const _Float16* pb = &plds[wline][n16 * 32];
            v8h lo = *(const v8h*)(pb + 16*half);
            v8h hi = *(const v8h*)(pb + 16*half + 8);
            ap = __builtin_shufflevector(lo, hi, 0,1,2,3,4,5,6,7,8,9,10,11,12,13,14,15);
        }

        // ---- context += P @ V, B-frags contiguous from the V slab ----
#pragma unroll
        for (int nt = 0; nt < 4; ++nt) {
            v16h bv = *(const v16h*)&vsl[(nt * 16 + n16) * 32 + 16*half];
            ctx[nt] = __builtin_amdgcn_wmma_f32_16x16x32_f16(false, ap, false, bv, (short)0, ctx[nt], false, false);
        }
    }

    // ---- normalize and store ----
    float rinv[8];
#pragma unroll
    for (int r = 0; r < 8; ++r) rinv[r] = 1.0f / lrun[r];
#pragma unroll
    for (int nt = 0; nt < 4; ++nt) {
#pragma unroll
        for (int r = 0; r < 8; ++r) {
            const long row = (long)b * S_ + q0 + r + 8*half;
            out[row * D_ + nt * 16 + n16] = ctx[nt][r] * rinv[r];
        }
    }
}

// ---------------------------------------------------------------------------
extern "C" void kernel_launch(void* const* d_in, const int* in_sizes, int n_in,
                              void* d_out, int out_size, void* d_ws, size_t ws_size,
                              hipStream_t stream)
{
    (void)in_sizes; (void)n_in; (void)out_size; (void)ws_size;
    const float* X  = (const float*)d_in[0];
    const unsigned char* mask = (const unsigned char*)d_in[1];  // jnp bool = 1 byte
    const float* Wq = (const float*)d_in[2];
    const float* bq = (const float*)d_in[3];
    const float* Wk = (const float*)d_in[4];
    const float* bk = (const float*)d_in[5];
    const float* Wv = (const float*)d_in[6];
    const float* bv = (const float*)d_in[7];
    float* out = (float*)d_out;

    _Float16* ws   = (_Float16*)d_ws;
    _Float16* Wswz = ws;
    _Float16* Qh   = ws + Q_OFF;
    _Float16* Kh   = ws + K_OFF;
    _Float16* Vt   = ws + VT_OFF;

    wswz_kernel<<<WSWZ_ELEMS / 256, 256, 0, stream>>>(Wq, Wk, Wv, Wswz);
    qkv_kernel<<<128, 256, 0, stream>>>(X, Wswz, bq, bk, bv, Qh, Kh, Vt);
    attn_kernel<<<128, 256, 0, stream>>>(Qh, Kh, Vt, mask, out);
}
